// EquivariantBlockPyG_77618648973486
// MI455X (gfx1250) — matrix operand order repaired and checked by reading, not compiled, not run on hardware
//
#include <hip/hip_runtime.h>

typedef unsigned short ushort_t;
typedef __attribute__((ext_vector_type(16))) __bf16        v16bf;
typedef __attribute__((ext_vector_type(8)))  float         v8f;
typedef __attribute__((ext_vector_type(4)))  unsigned int  u32x4;

#define HDIM 256
#define TWO_H 512
#define MTILE 32   // rows (edges or nodes) per block
#define NWARP 8    // 256 threads, wave32

union V16 { v16bf v; u32x4 u[2]; };

__device__ __forceinline__ ushort_t f2bf(float f) {
    unsigned u = __float_as_uint(f);
    u += 0x7FFFu + ((u >> 16) & 1u);
    return (ushort_t)(u >> 16);
}
__device__ __forceinline__ float silu_f(float x)    { return x / (1.0f + __expf(-x)); }
__device__ __forceinline__ float sigmoid_f(float x) { return 1.0f / (1.0f + __expf(-x)); }

// A fragment (16x32 bf16 tile) from row-major bf16 buffer, row stride `stride` elems.
__device__ __forceinline__ v16bf load_afrag(const ushort_t* base, int stride, int m0, int k0, int lane) {
    int m = lane & 15, half = lane >> 4;
    const ushort_t* p = base + (size_t)(m0 + m) * stride + k0 + half * 8;
    V16 r;
    r.u[0] = *(const u32x4*)(p);
    r.u[1] = *(const u32x4*)(p + 16);
    return r.v;
}
// B fragment (32x16 bf16 tile) from transposed weights Wt[outCol][K] bf16.
__device__ __forceinline__ v16bf load_bfrag(const ushort_t* wt, int K, int cb, int k0, int lane) {
    int n = lane & 15, half = lane >> 4;
    const ushort_t* p = wt + (size_t)(cb + n) * K + k0 + half * 8;
    V16 r;
    r.u[0] = *(const u32x4*)(p);
    r.u[1] = *(const u32x4*)(p + 16);
    return r.v;
}
#define WMMA_BF16(A, B, C) \
    __builtin_amdgcn_wmma_f32_16x16x32_bf16(false, (A), false, (B), (short)0, (C), false, false)

// ---------------- prep kernels ----------------
__global__ void k_f2bf_arr(const float* __restrict__ in, ushort_t* __restrict__ out, int n) {
    int i = blockIdx.x * blockDim.x + threadIdx.x;
    if (i < n) out[i] = f2bf(in[i]);
}
// in: f32 [K][Nc]  ->  out: bf16 [Nc][K]
__global__ void k_transpose_bf16(const float* __restrict__ in, ushort_t* __restrict__ out, int K, int Nc) {
    int i = blockIdx.x * blockDim.x + threadIdx.x;
    if (i >= K * Nc) return;
    int n = i / K, k = i - n * K;
    out[(size_t)n * K + k] = f2bf(in[(size_t)k * Nc + n]);
}
__global__ void k_coord2diff(const float* __restrict__ x, const int* __restrict__ eidx, int E_,
                             float* __restrict__ radial, float* __restrict__ cdiff) {
    int e = blockIdx.x * blockDim.x + threadIdx.x;
    if (e >= E_) return;
    int r = eidx[e], c = eidx[E_ + e];
    float d0 = x[r * 3 + 0] - x[c * 3 + 0];
    float d1 = x[r * 3 + 1] - x[c * 3 + 1];
    float d2 = x[r * 3 + 2] - x[c * 3 + 2];
    float rad = d0 * d0 + d1 * d1 + d2 * d2;
    float inv = 1.0f / (sqrtf(rad + 1e-8f) + 1.0f);
    radial[e] = rad;
    cdiff[e * 3 + 0] = d0 * inv;
    cdiff[e * 3 + 1] = d1 * inv;
    cdiff[e * 3 + 2] = d2 * inv;
}

// ---------------- fused edge kernel ----------------
__global__ __launch_bounds__(256) void edge_kernel(
    const ushort_t* __restrict__ hbf, const int* __restrict__ eidx, int E_,
    const ushort_t* __restrict__ w1t, const float* __restrict__ b1,
    const ushort_t* __restrict__ w2t, const float* __restrict__ b2,
    const float* __restrict__ attw, const float* __restrict__ attb,
    float* __restrict__ agg)
{
    __shared__ __align__(16) ushort_t shA[MTILE * TWO_H];   // 32KB; aliased as f32 m2 later
    __shared__ __align__(16) ushort_t shM1[MTILE * HDIM];   // 16KB
    __shared__ int shCol[MTILE], shRow[MTILE];
    __shared__ float shP[MTILE * 8];
    __shared__ float shAttv[MTILE];

    const int tid = threadIdx.x, lane = tid & 31, w = tid >> 5;
    const int e0 = blockIdx.x * MTILE;
    const int cb = w * 32;

    if (tid < MTILE) {
        shRow[tid] = eidx[e0 + tid];        // source j
        shCol[tid] = eidx[E_ + e0 + tid];   // target i
    }
    __syncthreads();

    { // gather concat(h[col], h[row]) -> shA[32][512]
        int r = tid >> 3, p = tid & 7;
        int node = (p < 4) ? shCol[r] : shRow[r];
        int off = (p & 3) * 64;
        const u32x4* src = (const u32x4*)(hbf + (size_t)node * HDIM + off);
        u32x4* dst = (u32x4*)(shA + r * TWO_H + ((p < 4) ? 0 : HDIM) + off);
        #pragma unroll
        for (int q = 0; q < 8; ++q) dst[q] = src[q];
    }
    __syncthreads();

    // GEMM1: [32x512] @ [512x256] slice, +b1, silu -> shM1 (bf16)
    v8f a00 = {}, a01 = {}, a10 = {}, a11 = {};
    for (int ks = 0; ks < 16; ++ks) {
        int k0 = ks * 32;
        v16bf fa0 = load_afrag(shA, TWO_H, 0,  k0, lane);
        v16bf fa1 = load_afrag(shA, TWO_H, 16, k0, lane);
        v16bf fb0 = load_bfrag(w1t, TWO_H, cb,      k0, lane);
        v16bf fb1 = load_bfrag(w1t, TWO_H, cb + 16, k0, lane);
        a00 = WMMA_BF16(fa0, fb0, a00);
        a01 = WMMA_BF16(fa0, fb1, a01);
        a10 = WMMA_BF16(fa1, fb0, a10);
        a11 = WMMA_BF16(fa1, fb1, a11);
    }
    {
        int n = lane & 15, half = lane >> 4;
        int c0 = cb + n, c1 = cb + 16 + n;
        float bb0 = b1[c0], bb1 = b1[c1];
        #pragma unroll
        for (int i = 0; i < 8; ++i) {
            int r0 = i + 8 * half;
            shM1[r0 * HDIM + c0]        = f2bf(silu_f(a00[i] + bb0));
            shM1[r0 * HDIM + c1]        = f2bf(silu_f(a01[i] + bb1));
            shM1[(16 + r0) * HDIM + c0] = f2bf(silu_f(a10[i] + bb0));
            shM1[(16 + r0) * HDIM + c1] = f2bf(silu_f(a11[i] + bb1));
        }
    }
    __syncthreads();

    // GEMM2: [32x256] @ [256x256] slice, +b2, silu -> shM2 (f32, aliases shA)
    v8f d00 = {}, d01 = {}, d10 = {}, d11 = {};
    for (int ks = 0; ks < 8; ++ks) {
        int k0 = ks * 32;
        v16bf fa0 = load_afrag(shM1, HDIM, 0,  k0, lane);
        v16bf fa1 = load_afrag(shM1, HDIM, 16, k0, lane);
        v16bf fb0 = load_bfrag(w2t, HDIM, cb,      k0, lane);
        v16bf fb1 = load_bfrag(w2t, HDIM, cb + 16, k0, lane);
        d00 = WMMA_BF16(fa0, fb0, d00);
        d01 = WMMA_BF16(fa0, fb1, d01);
        d10 = WMMA_BF16(fa1, fb0, d10);
        d11 = WMMA_BF16(fa1, fb1, d11);
    }
    float* shM2 = (float*)shA;  // 32*256*4 == 32*512*2 bytes
    {
        int n = lane & 15, half = lane >> 4;
        int c0 = cb + n, c1 = cb + 16 + n;
        float bb0 = b2[c0], bb1 = b2[c1];
        #pragma unroll
        for (int i = 0; i < 8; ++i) {
            int r0 = i + 8 * half;
            shM2[r0 * HDIM + c0]        = silu_f(d00[i] + bb0);
            shM2[r0 * HDIM + c1]        = silu_f(d01[i] + bb1);
            shM2[(16 + r0) * HDIM + c0] = silu_f(d10[i] + bb0);
            shM2[(16 + r0) * HDIM + c1] = silu_f(d11[i] + bb1);
        }
    }
    __syncthreads();

    // attention: att = sigmoid(m @ att_w + att_b)
    {
        int e = tid >> 3, c = tid & 7;
        float s = 0.f;
        #pragma unroll 8
        for (int j = 0; j < 32; ++j) s += shM2[e * HDIM + c * 32 + j] * attw[c * 32 + j];
        shP[e * 8 + c] = s;
    }
    __syncthreads();
    if (tid < MTILE) {
        float d = 0.f;
        #pragma unroll
        for (int c = 0; c < 8; ++c) d += shP[tid * 8 + c];
        shAttv[tid] = sigmoid_f(d + attb[0]);
    }
    __syncthreads();

    // scatter: agg[col] += m * att
    {
        int e = tid >> 3, c = tid & 7;
        float a = shAttv[e];
        float* dstp = agg + (size_t)shCol[e] * HDIM + c * 32;
        #pragma unroll 4
        for (int j = 0; j < 32; ++j)
            atomicAdd(&dstp[j], shM2[e * HDIM + c * 32 + j] * a);
    }
}

// ---------------- node kernel ----------------
__global__ __launch_bounds__(256) void node_kernel(
    const ushort_t* __restrict__ hbf_in, const float* __restrict__ agg,
    const ushort_t* __restrict__ w1t, const float* __restrict__ b1,
    const ushort_t* __restrict__ w2t, const float* __restrict__ b2,
    ushort_t* __restrict__ hbf_out, float* __restrict__ hf32_out)
{
    __shared__ __align__(16) ushort_t shA[MTILE * TWO_H];
    __shared__ __align__(16) ushort_t shM1[MTILE * HDIM];

    const int tid = threadIdx.x, lane = tid & 31, w = tid >> 5;
    const int n0 = blockIdx.x * MTILE;
    const int cb = w * 32;

    { // load concat(h, agg) -> shA
        int r = tid >> 3, p = tid & 7;
        int node = n0 + r;
        int off = (p & 3) * 64;
        if (p < 4) {
            const u32x4* src = (const u32x4*)(hbf_in + (size_t)node * HDIM + off);
            u32x4* dst = (u32x4*)(shA + r * TWO_H + off);
            #pragma unroll
            for (int q = 0; q < 8; ++q) dst[q] = src[q];
        } else {
            const float* src = agg + (size_t)node * HDIM + off;
            ushort_t* dst = shA + r * TWO_H + HDIM + off;
            #pragma unroll 8
            for (int q = 0; q < 64; ++q) dst[q] = f2bf(src[q]);
        }
    }
    __syncthreads();

    v8f a00 = {}, a01 = {}, a10 = {}, a11 = {};
    for (int ks = 0; ks < 16; ++ks) {
        int k0 = ks * 32;
        v16bf fa0 = load_afrag(shA, TWO_H, 0,  k0, lane);
        v16bf fa1 = load_afrag(shA, TWO_H, 16, k0, lane);
        v16bf fb0 = load_bfrag(w1t, TWO_H, cb,      k0, lane);
        v16bf fb1 = load_bfrag(w1t, TWO_H, cb + 16, k0, lane);
        a00 = WMMA_BF16(fa0, fb0, a00);
        a01 = WMMA_BF16(fa0, fb1, a01);
        a10 = WMMA_BF16(fa1, fb0, a10);
        a11 = WMMA_BF16(fa1, fb1, a11);
    }
    {
        int n = lane & 15, half = lane >> 4;
        int c0 = cb + n, c1 = cb + 16 + n;
        float bb0 = b1[c0], bb1 = b1[c1];
        #pragma unroll
        for (int i = 0; i < 8; ++i) {
            int r0 = i + 8 * half;
            shM1[r0 * HDIM + c0]        = f2bf(silu_f(a00[i] + bb0));
            shM1[r0 * HDIM + c1]        = f2bf(silu_f(a01[i] + bb1));
            shM1[(16 + r0) * HDIM + c0] = f2bf(silu_f(a10[i] + bb0));
            shM1[(16 + r0) * HDIM + c1] = f2bf(silu_f(a11[i] + bb1));
        }
    }
    __syncthreads();

    v8f d00 = {}, d01 = {}, d10 = {}, d11 = {};
    for (int ks = 0; ks < 8; ++ks) {
        int k0 = ks * 32;
        v16bf fa0 = load_afrag(shM1, HDIM, 0,  k0, lane);
        v16bf fa1 = load_afrag(shM1, HDIM, 16, k0, lane);
        v16bf fb0 = load_bfrag(w2t, HDIM, cb,      k0, lane);
        v16bf fb1 = load_bfrag(w2t, HDIM, cb + 16, k0, lane);
        d00 = WMMA_BF16(fa0, fb0, d00);
        d01 = WMMA_BF16(fa0, fb1, d01);
        d10 = WMMA_BF16(fa1, fb0, d10);
        d11 = WMMA_BF16(fa1, fb1, d11);
    }
    {
        int n = lane & 15, half = lane >> 4;
        int c0 = cb + n, c1 = cb + 16 + n;
        float bb0 = b2[c0], bb1 = b2[c1];
        bool wf = (hf32_out != nullptr);
        #pragma unroll
        for (int i = 0; i < 8; ++i) {
            int r0 = i + 8 * half;
            float v0 = d00[i] + bb0, v1 = d01[i] + bb1;
            float v2 = d10[i] + bb0, v3 = d11[i] + bb1;
            size_t g0 = (size_t)(n0 + r0) * HDIM + c0;
            size_t g1 = (size_t)(n0 + r0) * HDIM + c1;
            size_t g2 = (size_t)(n0 + 16 + r0) * HDIM + c0;
            size_t g3 = (size_t)(n0 + 16 + r0) * HDIM + c1;
            hbf_out[g0] = f2bf(v0); hbf_out[g1] = f2bf(v1);
            hbf_out[g2] = f2bf(v2); hbf_out[g3] = f2bf(v3);
            if (wf) { hf32_out[g0] = v0; hf32_out[g1] = v1; hf32_out[g2] = v2; hf32_out[g3] = v3; }
        }
    }
}

// ---------------- coord (EquivariantUpdate) kernel ----------------
__global__ __launch_bounds__(256) void coord_kernel(
    const ushort_t* __restrict__ hbf, const int* __restrict__ eidx, int E_,
    const ushort_t* __restrict__ w1t, const float* __restrict__ cw1_full, const float* __restrict__ b1,
    const ushort_t* __restrict__ w2t, const float* __restrict__ b2,
    const float* __restrict__ cw3,
    const float* __restrict__ radial, const float* __restrict__ eattr,
    const float* __restrict__ cdiff, float* __restrict__ xout)
{
    __shared__ __align__(16) ushort_t shA[MTILE * TWO_H];
    __shared__ __align__(16) ushort_t shM1[MTILE * HDIM];
    __shared__ int shCol[MTILE], shRow[MTILE];
    __shared__ float shRad[MTILE], shEA[MTILE];
    __shared__ float shP[MTILE * 8];

    const int tid = threadIdx.x, lane = tid & 31, w = tid >> 5;
    const int e0 = blockIdx.x * MTILE;
    const int cb = w * 32;

    if (tid < MTILE) {
        shRow[tid] = eidx[e0 + tid];
        shCol[tid] = eidx[E_ + e0 + tid];
        shRad[tid] = radial[e0 + tid];
        shEA[tid]  = eattr[e0 + tid];
    }
    __syncthreads();

    { // gather concat(h[col], h[row])
        int r = tid >> 3, p = tid & 7;
        int node = (p < 4) ? shCol[r] : shRow[r];
        int off = (p & 3) * 64;
        const u32x4* src = (const u32x4*)(hbf + (size_t)node * HDIM + off);
        u32x4* dst = (u32x4*)(shA + r * TWO_H + ((p < 4) ? 0 : HDIM) + off);
        #pragma unroll
        for (int q = 0; q < 8; ++q) dst[q] = src[q];
    }
    __syncthreads();

    v8f a00 = {}, a01 = {}, a10 = {}, a11 = {};
    for (int ks = 0; ks < 16; ++ks) {
        int k0 = ks * 32;
        v16bf fa0 = load_afrag(shA, TWO_H, 0,  k0, lane);
        v16bf fa1 = load_afrag(shA, TWO_H, 16, k0, lane);
        v16bf fb0 = load_bfrag(w1t, TWO_H, cb,      k0, lane);
        v16bf fb1 = load_bfrag(w1t, TWO_H, cb + 16, k0, lane);
        a00 = WMMA_BF16(fa0, fb0, a00);
        a01 = WMMA_BF16(fa0, fb1, a01);
        a10 = WMMA_BF16(fa1, fb0, a10);
        a11 = WMMA_BF16(fa1, fb1, a11);
    }
    { // + radial*cw1[512,:] + edge_attr*cw1[513,:] + b1, silu
        int n = lane & 15, half = lane >> 4;
        int c0 = cb + n, c1 = cb + 16 + n;
        float wr0 = cw1_full[512 * HDIM + c0], wr1 = cw1_full[512 * HDIM + c1];
        float we0 = cw1_full[513 * HDIM + c0], we1 = cw1_full[513 * HDIM + c1];
        float bb0 = b1[c0], bb1 = b1[c1];
        #pragma unroll
        for (int i = 0; i < 8; ++i) {
            int r0 = i + 8 * half, r1 = 16 + r0;
            float x0 = a00[i] + shRad[r0] * wr0 + shEA[r0] * we0 + bb0;
            float x1 = a01[i] + shRad[r0] * wr1 + shEA[r0] * we1 + bb1;
            float x2 = a10[i] + shRad[r1] * wr0 + shEA[r1] * we0 + bb0;
            float x3 = a11[i] + shRad[r1] * wr1 + shEA[r1] * we1 + bb1;
            shM1[r0 * HDIM + c0] = f2bf(silu_f(x0));
            shM1[r0 * HDIM + c1] = f2bf(silu_f(x1));
            shM1[r1 * HDIM + c0] = f2bf(silu_f(x2));
            shM1[r1 * HDIM + c1] = f2bf(silu_f(x3));
        }
    }
    __syncthreads();

    v8f d00 = {}, d01 = {}, d10 = {}, d11 = {};
    for (int ks = 0; ks < 8; ++ks) {
        int k0 = ks * 32;
        v16bf fa0 = load_afrag(shM1, HDIM, 0,  k0, lane);
        v16bf fa1 = load_afrag(shM1, HDIM, 16, k0, lane);
        v16bf fb0 = load_bfrag(w2t, HDIM, cb,      k0, lane);
        v16bf fb1 = load_bfrag(w2t, HDIM, cb + 16, k0, lane);
        d00 = WMMA_BF16(fa0, fb0, d00);
        d01 = WMMA_BF16(fa0, fb1, d01);
        d10 = WMMA_BF16(fa1, fb0, d10);
        d11 = WMMA_BF16(fa1, fb1, d11);
    }
    float* shM2 = (float*)shA;
    {
        int n = lane & 15, half = lane >> 4;
        int c0 = cb + n, c1 = cb + 16 + n;
        float bb0 = b2[c0], bb1 = b2[c1];
        #pragma unroll
        for (int i = 0; i < 8; ++i) {
            int r0 = i + 8 * half;
            shM2[r0 * HDIM + c0]        = silu_f(d00[i] + bb0);
            shM2[r0 * HDIM + c1]        = silu_f(d01[i] + bb1);
            shM2[(16 + r0) * HDIM + c0] = silu_f(d10[i] + bb0);
            shM2[(16 + r0) * HDIM + c1] = silu_f(d11[i] + bb1);
        }
    }
    __syncthreads();

    { // s = t @ cw3
        int e = tid >> 3, c = tid & 7;
        float s = 0.f;
        #pragma unroll 8
        for (int j = 0; j < 32; ++j) s += shM2[e * HDIM + c * 32 + j] * cw3[c * 32 + j];
        shP[e * 8 + c] = s;
    }
    __syncthreads();
    if (tid < MTILE) {
        float s = 0.f;
        #pragma unroll
        for (int c = 0; c < 8; ++c) s += shP[tid * 8 + c];
        int ci = shCol[tid];
        int e = e0 + tid;
        atomicAdd(&xout[ci * 3 + 0], cdiff[e * 3 + 0] * s);
        atomicAdd(&xout[ci * 3 + 1], cdiff[e * 3 + 1] * s);
        atomicAdd(&xout[ci * 3 + 2], cdiff[e * 3 + 2] * s);
    }
}

// ---------------- host launch ----------------
extern "C" void kernel_launch(void* const* d_in, const int* in_sizes, int n_in,
                              void* d_out, int out_size, void* d_ws, size_t ws_size,
                              hipStream_t stream) {
    (void)in_sizes; (void)n_in; (void)out_size; (void)ws_size;
    const int N = 8192, E_ = 262144, L = 2;

    const float* h_in  = (const float*)d_in[0];
    const float* x_in  = (const float*)d_in[1];
    const float* eattr = (const float*)d_in[2];
    const int*   eidx  = (const int*)  d_in[3];
    const float* ew1 = (const float*)d_in[4];
    const float* eb1 = (const float*)d_in[5];
    const float* ew2 = (const float*)d_in[6];
    const float* eb2 = (const float*)d_in[7];
    const float* attw = (const float*)d_in[8];
    const float* attb = (const float*)d_in[9];
    const float* nw1 = (const float*)d_in[10];
    const float* nb1 = (const float*)d_in[11];
    const float* nw2 = (const float*)d_in[12];
    const float* nb2 = (const float*)d_in[13];
    const float* cw1 = (const float*)d_in[14];
    const float* cb1 = (const float*)d_in[15];
    const float* cw2 = (const float*)d_in[16];
    const float* cb2 = (const float*)d_in[17];
    const float* cw3 = (const float*)d_in[18];

    char* ws = (char*)d_ws;
    size_t off = 0;
    auto wsalloc = [&](size_t bytes) -> char* {
        char* p = ws + off; off += (bytes + 255) & ~(size_t)255; return p;
    };
    ushort_t* hbfA   = (ushort_t*)wsalloc((size_t)N * HDIM * 2);
    ushort_t* hbfB   = (ushort_t*)wsalloc((size_t)N * HDIM * 2);
    float*    agg    = (float*)   wsalloc((size_t)N * HDIM * 4);
    float*    radial = (float*)   wsalloc((size_t)E_ * 4);
    float*    cdiff  = (float*)   wsalloc((size_t)E_ * 12);
    ushort_t* ew1t   = (ushort_t*)wsalloc((size_t)2 * 256 * 512 * 2);
    ushort_t* ew2t   = (ushort_t*)wsalloc((size_t)2 * 256 * 256 * 2);
    ushort_t* nw1t   = (ushort_t*)wsalloc((size_t)2 * 256 * 512 * 2);
    ushort_t* nw2t   = (ushort_t*)wsalloc((size_t)2 * 256 * 256 * 2);
    ushort_t* cw1t   = (ushort_t*)wsalloc((size_t)256 * 512 * 2);
    ushort_t* cw2t   = (ushort_t*)wsalloc((size_t)256 * 256 * 2);

    float* out_h = (float*)d_out;
    float* out_x = out_h + (size_t)N * HDIM;

    dim3 blk(256);
    k_f2bf_arr<<<(N * HDIM) / 256, blk, 0, stream>>>(h_in, hbfA, N * HDIM);
    for (int l = 0; l < L; ++l) {
        k_transpose_bf16<<<(512 * 256) / 256, blk, 0, stream>>>(ew1 + (size_t)l * 512 * 256, ew1t + (size_t)l * 256 * 512, 512, 256);
        k_transpose_bf16<<<(256 * 256) / 256, blk, 0, stream>>>(ew2 + (size_t)l * 256 * 256, ew2t + (size_t)l * 256 * 256, 256, 256);
        k_transpose_bf16<<<(512 * 256) / 256, blk, 0, stream>>>(nw1 + (size_t)l * 512 * 256, nw1t + (size_t)l * 256 * 512, 512, 256);
        k_transpose_bf16<<<(256 * 256) / 256, blk, 0, stream>>>(nw2 + (size_t)l * 256 * 256, nw2t + (size_t)l * 256 * 256, 256, 256);
    }
    k_transpose_bf16<<<(512 * 256) / 256, blk, 0, stream>>>(cw1, cw1t, 512, 256);
    k_transpose_bf16<<<(256 * 256) / 256, blk, 0, stream>>>(cw2, cw2t, 256, 256);
    k_coord2diff<<<E_ / 256, blk, 0, stream>>>(x_in, eidx, E_, radial, cdiff);
    hipMemcpyAsync(out_x, x_in, (size_t)N * 3 * 4, hipMemcpyDeviceToDevice, stream);

    ushort_t* hcur = hbfA;
    ushort_t* hnext = hbfB;
    for (int l = 0; l < L; ++l) {
        hipMemsetAsync(agg, 0, (size_t)N * HDIM * 4, stream);
        edge_kernel<<<E_ / MTILE, blk, 0, stream>>>(hcur, eidx, E_,
            ew1t + (size_t)l * 256 * 512, eb1 + (size_t)l * 256,
            ew2t + (size_t)l * 256 * 256, eb2 + (size_t)l * 256,
            attw + (size_t)l * 256, attb + l, agg);
        node_kernel<<<N / MTILE, blk, 0, stream>>>(hcur, agg,
            nw1t + (size_t)l * 256 * 512, nb1 + (size_t)l * 256,
            nw2t + (size_t)l * 256 * 256, nb2 + (size_t)l * 256,
            hnext, (l == L - 1) ? out_h : nullptr);
        ushort_t* t = hcur; hcur = hnext; hnext = t;
    }
    coord_kernel<<<E_ / MTILE, blk, 0, stream>>>(hcur, eidx, E_,
        cw1t, cw1, cb1, cw2t, cb2, cw3, radial, eattr, cdiff, out_x);
}
